// TextEncoder_31945966748146
// MI455X (gfx1250) — matrix operand order repaired and checked by reading, not compiled, not run on hardware
//
#include <hip/hip_runtime.h>
#include <hip/hip_bf16.h>
#include <math.h>

// ---------------------------------------------------------------------------
// TextEncoder forward for MI455X (gfx1250): bf16 WMMA GEMMs (async-LDS
// double-buffered) + flash attention.
// L=6, H=8, D=512, FF=2048, V=32000, B=4, N=2048, SCALE = D^-0.5
// ---------------------------------------------------------------------------

typedef __bf16 v16bf __attribute__((ext_vector_type(16)));
typedef float  v8f   __attribute__((ext_vector_type(8)));

static constexpr int   Hh = 8, Dd = 512, Bb = 4, Nn = 2048;
static constexpr float SCALE_QK = 0.044194173824159216f; // 512^-0.5

union Frag16 { v16bf v; uint4 u[2]; __bf16 e[16]; };
union Pack8  { uint4 u;  __bf16 e[8]; };

// CDNA5 async global->LDS copy of 16 bytes per lane (ASYNCcnt-tracked).
__device__ __forceinline__ void async_b128(unsigned lds_off, const void* gaddr) {
  asm volatile("global_load_async_to_lds_b128 %0, %1, off"
               :: "v"(lds_off), "v"(gaddr) : "memory");
}

// ------------------------------ embedding + PE -----------------------------
__global__ void embed_pe_kernel(const int* __restrict__ tokens,
                                const float* __restrict__ emb,
                                float* __restrict__ x, int total) {
  int idx = blockIdx.x * blockDim.x + threadIdx.x;
  if (idx >= total) return;
  int d  = idx % Dd;
  int bn = idx / Dd;
  int n  = bn % Nn;
  int tok = tokens[bn];
  float ang = (float)n * powf(10000.0f, -(float)(d & ~1) / (float)Dd);
  float pe  = (d & 1) ? cosf(ang) : sinf(ang);
  x[idx] = emb[(size_t)tok * Dd + d] + pe;
}

// ------------------------------ conversions --------------------------------
__global__ void cvt_bf16_kernel(const float* __restrict__ s,
                                __bf16* __restrict__ d, int n) {
  int i = blockIdx.x * 256 + threadIdx.x;
  if (i < n) d[i] = (__bf16)s[i];
}

// W[K,N] fp32 -> Wt[N,K] bf16 (row-major transposed for contiguous B-frags)
__global__ void tcvt_bf16_kernel(const float* __restrict__ W,
                                 __bf16* __restrict__ Wt, int K, int N) {
  int i = blockIdx.x * 256 + threadIdx.x;
  if (i < K * N) {
    int n = i / K, k = i % K;
    Wt[i] = (__bf16)W[(size_t)k * N + n];
  }
}

// ------------------------------ GEMM (bf16 WMMA) ---------------------------
// C[M,N] = act(A[M,K] @ Bt[N,K]^T + bias), act: 0=none, 1=exact GELU
// Staging: double-buffered LDS tiles filled by global_load_async_to_lds_b128.
#define GBM 128
#define GBN 64
#define GBK 32
#define LDA 40
#define LDB 40

__global__ __launch_bounds__(256) void gemm_bf16_wmma(
    const __bf16* __restrict__ A, const __bf16* __restrict__ Bt,
    const float* __restrict__ bias, float* __restrict__ C,
    int M, int N, int K, int act) {
  __shared__ __bf16 As[2][GBM * LDA];
  __shared__ __bf16 Bs[2][GBN * LDB];
  int tid = threadIdx.x;
  int lane = tid & 31, w = tid >> 5;
  int wm = (w & 3) * 32, wn = (w >> 2) * 32;
  int m0 = blockIdx.y * GBM, n0 = blockIdx.x * GBN;

  // per-thread staging coordinates (3 async b128 per thread per tile)
  int r0 = tid >> 2,          c0 = (tid & 3) * 8;
  int r1 = (tid + 256) >> 2,  c1 = ((tid + 256) & 3) * 8;
  int rb = tid >> 2,          cb = (tid & 3) * 8;
  const __bf16* gA0 = &A[(size_t)(m0 + r0) * K + c0];
  const __bf16* gA1 = &A[(size_t)(m0 + r1) * K + c1];
  const __bf16* gB  = &Bt[(size_t)(n0 + rb) * K + cb];
  unsigned lA0[2], lA1[2], lB[2];
  for (int bfi = 0; bfi < 2; ++bfi) {
    lA0[bfi] = (unsigned)(uintptr_t)&As[bfi][r0 * LDA + c0];
    lA1[bfi] = (unsigned)(uintptr_t)&As[bfi][r1 * LDA + c1];
    lB[bfi]  = (unsigned)(uintptr_t)&Bs[bfi][rb * LDB + cb];
  }

  v8f acc[2][2] = {};
  int am = lane & 15;
  int half = lane >> 4;
  int nk = K / GBK;

  // prologue: issue tile 0 into buffer 0
  async_b128(lA0[0], gA0);
  async_b128(lA1[0], gA1);
  async_b128(lB[0],  gB);

  for (int kt = 0; kt < nk; ++kt) {
    int cur = kt & 1;
    if (kt + 1 < nk) {
      int k0 = (kt + 1) * GBK;
      async_b128(lA0[1 - cur], gA0 + k0);
      async_b128(lA1[1 - cur], gA1 + k0);
      async_b128(lB[1 - cur],  gB + k0);
      asm volatile("s_wait_asynccnt 0x3" ::: "memory");  // current tile landed
    } else {
      asm volatile("s_wait_asynccnt 0x0" ::: "memory");
    }
    __syncthreads();  // tile visible to all waves

    Frag16 af[2], bf_[2];
    int ac = half * 8;   // A frag: k = (i/8)*16 + half*8 + i%8
    for (int mi = 0; mi < 2; ++mi) {
      const __bf16* p = &As[cur][(wm + mi * 16 + am) * LDA];
      af[mi].u[0] = *(const uint4*)&p[ac];
      af[mi].u[1] = *(const uint4*)&p[ac + 16];
    }
    int bc = half * 8;   // B frag: k = (i/8)*16 + half*8 + i%8 (Bt stored [n][k])
    for (int ni = 0; ni < 2; ++ni) {
      const __bf16* p = &Bs[cur][(wn + ni * 16 + am) * LDB + bc];
      bf_[ni].u[0] = *(const uint4*)&p[0];
      bf_[ni].u[1] = *(const uint4*)&p[16];
    }
#pragma unroll
    for (int mi = 0; mi < 2; ++mi)
#pragma unroll
      for (int ni = 0; ni < 2; ++ni)
        acc[mi][ni] = __builtin_amdgcn_wmma_f32_16x16x32_bf16(
            false, af[mi].v, false, bf_[ni].v, (short)0, acc[mi][ni], false, false);
    __syncthreads();  // all waves done reading -> buffer reusable next+1 iter
  }

  // epilogue: C layout m = r + 8*half, n = lane&15
  for (int mi = 0; mi < 2; ++mi)
    for (int ni = 0; ni < 2; ++ni) {
      int ng = n0 + wn + ni * 16 + am;
      float bv = bias[ng];
      for (int r = 0; r < 8; ++r) {
        int mg = m0 + wm + mi * 16 + 8 * half + r;
        float v = acc[mi][ni][r] + bv;
        if (act) v = 0.5f * v * (1.0f + erff(v * 0.70710678118f));
        C[(size_t)mg * N + ng] = v;
      }
    }
}

// ------------------------------ flash attention ----------------------------
// qkv: [B,N,3D] fp32 ; out: [B,N,D] fp32. One wave = 16 q rows, block = 128.
#define KLD 72
#define VLD 40
#define PLD 40

__global__ __launch_bounds__(256) void flash_attn_kernel(
    const float* __restrict__ qkv, float* __restrict__ out) {
  __shared__ __bf16 Ks[32 * KLD];        // [key][dh]
  __shared__ __bf16 Vt[64 * VLD];        // [dh][key] (transposed)
  __shared__ __bf16 Ps[8 * 16 * PLD];    // per-wave P tile [m][key]
  int tid = threadIdx.x, lane = tid & 31, w = tid >> 5;
  int bh = blockIdx.y, b = bh / Hh, h = bh % Hh;
  int q0 = blockIdx.x * 128 + w * 16;
  const size_t rs = 3 * Dd;              // 1536 floats per token row
  int m = lane & 15, half = lane >> 4;

  // Q fragments for the two 32-wide dh chunks, pre-scaled by D^-0.5
  Frag16 qf[2];
  {
    const float* qrow = qkv + ((size_t)(b * Nn + q0 + m)) * rs + h * 64;
    for (int c = 0; c < 2; ++c) {
      int base = c * 32 + half * 8;
      for (int i = 0; i < 8; ++i) qf[c].e[i]     = (__bf16)(qrow[base + i]      * SCALE_QK);
      for (int i = 0; i < 8; ++i) qf[c].e[8 + i] = (__bf16)(qrow[base + 16 + i] * SCALE_QK);
    }
  }

  float mrow[8], lrow[8];
  v8f oacc[4] = {};
  for (int r = 0; r < 8; ++r) { mrow[r] = -1e30f; lrow[r] = 0.0f; }

  for (int kc = 0; kc < Nn; kc += 32) {
    // cooperative stage of 32-key K chunk (row-major) and V chunk (transposed)
    {
      int idx = tid * 8;                 // 2048 bf16 elements, 8/thread
      int key = idx >> 6, dh = idx & 63;
      const float* krow = qkv + ((size_t)(b * Nn + kc + key)) * rs + Dd + h * 64 + dh;
      Pack8 t8;
      for (int i = 0; i < 8; ++i) t8.e[i] = (__bf16)krow[i];
      *(uint4*)&Ks[key * KLD + dh] = t8.u;
      const float* vrow = qkv + ((size_t)(b * Nn + kc + key)) * rs + 2 * Dd + h * 64 + dh;
      for (int i = 0; i < 8; ++i) Vt[(dh + i) * VLD + key] = (__bf16)vrow[i];
    }
    __syncthreads();

    // scores: Q(16x64) @ K^T(64x32) -> two 16x16 f32 tiles
    v8f sc[2] = {};
#pragma unroll
    for (int sub = 0; sub < 2; ++sub)
#pragma unroll
      for (int c = 0; c < 2; ++c) {
        Frag16 kf;
        const __bf16* p = &Ks[(sub * 16 + m) * KLD + c * 32 + half * 8];
        kf.u[0] = *(const uint4*)&p[0];
        kf.u[1] = *(const uint4*)&p[16];
        sc[sub] = __builtin_amdgcn_wmma_f32_16x16x32_bf16(
            false, qf[c].v, false, kf.v, (short)0, sc[sub], false, false);
      }

    // online softmax (rows striped: row = r + 8*half across 16-lane groups)
    __bf16* pw = &Ps[w * 16 * PLD];
    for (int r = 0; r < 8; ++r) {
      float cm = fmaxf(sc[0][r], sc[1][r]);
      for (int off = 8; off; off >>= 1) cm = fmaxf(cm, __shfl_xor(cm, off, 16));
      float nm = fmaxf(mrow[r], cm);
      float sc_old = __expf(mrow[r] - nm);
      float p0 = __expf(sc[0][r] - nm);
      float p1 = __expf(sc[1][r] - nm);
      float ps = p0 + p1;
      for (int off = 8; off; off >>= 1) ps += __shfl_xor(ps, off, 16);
      lrow[r] = lrow[r] * sc_old + ps;
      mrow[r] = nm;
      for (int t = 0; t < 4; ++t) oacc[t][r] *= sc_old;
      int mr = r + 8 * half;
      pw[mr * PLD + 0 * 16 + m] = (__bf16)p0;
      pw[mr * PLD + 1 * 16 + m] = (__bf16)p1;
    }
    asm volatile("s_wait_dscnt 0" ::: "memory");  // P tile LDS RAW within wave
    __builtin_amdgcn_wave_barrier();

    // O += P(16x32) @ V(32x64)
    Frag16 pf;
    {
      const __bf16* p = &pw[m * PLD + half * 8];
      pf.u[0] = *(const uint4*)&p[0];
      pf.u[1] = *(const uint4*)&p[16];
    }
#pragma unroll
    for (int t = 0; t < 4; ++t) {
      Frag16 vf;
      const __bf16* p = &Vt[(t * 16 + m) * VLD + half * 8];
      vf.u[0] = *(const uint4*)&p[0];
      vf.u[1] = *(const uint4*)&p[16];
      oacc[t] = __builtin_amdgcn_wmma_f32_16x16x32_bf16(
          false, pf.v, false, vf.v, (short)0, oacc[t], false, false);
    }
    __syncthreads();
  }

  // write: out[b, q, h*64 + dh]  (== attn transposed back to [B,N,D])
  for (int t = 0; t < 4; ++t)
    for (int r = 0; r < 8; ++r) {
      int mr = r + 8 * half;
      out[((size_t)(b * Nn + q0 + mr)) * Dd + h * 64 + t * 16 + m] =
          oacc[t][r] / lrow[r];
    }
}

// ------------------------------ residual + LayerNorm -----------------------
__global__ __launch_bounds__(256) void add_ln_kernel(
    const float* __restrict__ a, const float* __restrict__ bres,
    const float* __restrict__ g, const float* __restrict__ beta,
    float* __restrict__ y) {
  int row = blockIdx.x, tid = threadIdx.x;
  const float* pa = a + (size_t)row * Dd;
  const float* pb = bres + (size_t)row * Dd;
  float v0 = pa[tid] + pb[tid];
  float v1 = pa[tid + 256] + pb[tid + 256];
  float s = v0 + v1, s2 = v0 * v0 + v1 * v1;
  __shared__ float sh1[8], sh2[8];
  for (int off = 16; off; off >>= 1) {
    s += __shfl_xor(s, off, 32);
    s2 += __shfl_xor(s2, off, 32);
  }
  int w = tid >> 5, lane = tid & 31;
  if (lane == 0) { sh1[w] = s; sh2[w] = s2; }
  __syncthreads();
  if (w == 0) {
    float t1 = (lane < 8) ? sh1[lane] : 0.0f;
    float t2 = (lane < 8) ? sh2[lane] : 0.0f;
    for (int off = 4; off; off >>= 1) {
      t1 += __shfl_xor(t1, off, 32);
      t2 += __shfl_xor(t2, off, 32);
    }
    if (lane == 0) { sh1[0] = t1; sh2[0] = t2; }
  }
  __syncthreads();
  float mean = sh1[0] * (1.0f / Dd);
  float var  = sh2[0] * (1.0f / Dd) - mean * mean;
  float inv  = rsqrtf(var + 1e-5f);
  y[(size_t)row * Dd + tid]       = (v0 - mean) * inv * g[tid] + beta[tid];
  y[(size_t)row * Dd + tid + 256] = (v1 - mean) * inv * g[tid + 256] + beta[tid + 256];
}

// ------------------------------ driver -------------------------------------
extern "C" void kernel_launch(void* const* d_in, const int* in_sizes, int n_in,
                              void* d_out, int out_size, void* d_ws, size_t ws_size,
                              hipStream_t stream) {
  (void)in_sizes; (void)n_in; (void)out_size; (void)ws_size;
  const int*   tokens = (const int*)d_in[0];
  const float* emb    = (const float*)d_in[1];
  const float* Wqkv   = (const float*)d_in[2];
  const float* bqkv   = (const float*)d_in[3];
  const float* g1     = (const float*)d_in[4];
  const float* b1     = (const float*)d_in[5];
  const float* W1     = (const float*)d_in[6];
  const float* bf1    = (const float*)d_in[7];
  const float* W2     = (const float*)d_in[8];
  const float* bf2    = (const float*)d_in[9];
  const float* g2     = (const float*)d_in[10];
  const float* b2     = (const float*)d_in[11];

  char* ws = (char*)d_ws;
  const int    M  = Bb * Nn;                        // 8192
  const size_t XB = (size_t)M * Dd * sizeof(float); // 16.78 MB
  float*  x    = (float*)(ws + 0);
  float*  y    = (float*)(ws + XB);
  float*  attn = (float*)(ws + 2 * XB);
  float*  obuf = (float*)(ws + 3 * XB);
  float*  qkv  = (float*)(ws + 4 * XB);             // 3*XB
  float*  hbuf = (float*)(ws + 7 * XB);             // 4*XB
  __bf16* xb   = (__bf16*)(ws + 11 * XB);           // up to 2*XB
  __bf16* wT   = (__bf16*)(ws + 13 * XB);           // <= 2 MB

  embed_pe_kernel<<<(M * Dd + 255) / 256, 256, 0, stream>>>(tokens, emb, x, M * Dd);

  for (int l = 0; l < 6; ++l) {
    // ---- QKV projection: [8192,512] @ [512,1536] + bias ----
    cvt_bf16_kernel<<<(M * 512 + 255) / 256, 256, 0, stream>>>(x, xb, M * 512);
    tcvt_bf16_kernel<<<(512 * 1536 + 255) / 256, 256, 0, stream>>>(
        Wqkv + (size_t)l * 512 * 1536, wT, 512, 1536);
    gemm_bf16_wmma<<<dim3(1536 / GBN, M / GBM), 256, 0, stream>>>(
        xb, wT, bqkv + (size_t)l * 1536, qkv, M, 1536, 512, 0);

    // ---- attention (flash, per (b,h)) ----
    flash_attn_kernel<<<dim3(Nn / 128, Bb * Hh), 256, 0, stream>>>(qkv, attn);

    // ---- y = LN(attn + x) ----
    add_ln_kernel<<<M, 256, 0, stream>>>(attn, x, g1 + (size_t)l * 512,
                                         b1 + (size_t)l * 512, y);

    // ---- FF1: gelu([8192,512] @ [512,2048] + bias) ----
    cvt_bf16_kernel<<<(M * 512 + 255) / 256, 256, 0, stream>>>(y, xb, M * 512);
    tcvt_bf16_kernel<<<(512 * 2048 + 255) / 256, 256, 0, stream>>>(
        W1 + (size_t)l * 512 * 2048, wT, 512, 2048);
    gemm_bf16_wmma<<<dim3(2048 / GBN, M / GBM), 256, 0, stream>>>(
        xb, wT, bf1 + (size_t)l * 2048, hbuf, M, 2048, 512, 1);

    // ---- FF2: gelu([8192,2048] @ [2048,512] + bias) ----
    cvt_bf16_kernel<<<(M * 2048 + 255) / 256, 256, 0, stream>>>(hbuf, xb, M * 2048);
    tcvt_bf16_kernel<<<(2048 * 512 + 255) / 256, 256, 0, stream>>>(
        W2 + (size_t)l * 2048 * 512, wT, 2048, 512);
    gemm_bf16_wmma<<<dim3(512 / GBN, M / GBM), 256, 0, stream>>>(
        xb, wT, bf2 + (size_t)l * 512, obuf, M, 512, 2048, 1);

    // ---- x = LN(o + y) ----
    add_ln_kernel<<<M, 256, 0, stream>>>(obuf, y, g2 + (size_t)l * 512,
                                         b2 + (size_t)l * 512, x);
  }

  hipMemcpyAsync(d_out, x, XB, hipMemcpyDeviceToDevice, stream);
}
